// EnhancedPointNet2Encoder_5368709120556
// MI455X (gfx1250) — matrix operand order repaired and checked by reading, not compiled, not run on hardware
//
#include <hip/hip_runtime.h>
#include <stdint.h>

typedef __attribute__((ext_vector_type(16))) _Float16 v16h;
typedef __attribute__((ext_vector_type(8)))  _Float16 v8h;
typedef __attribute__((ext_vector_type(8)))  float    v8f;

// ---------------------------------------------------------------------------
// Weight convert + swizzle: fp32 (Kin x N) -> fp16 pre-swizzled into the WMMA
// B-operand lane layout so each lane's 16 halves are contiguous (32B):
//   bswz[(((kb*NT + tn)*32 + lane)*16 + i] = W[kb*32 + (lane>=16)*8 + (i&7) + (i>>3)*16][tn*16 + (lane&15)]
// ---------------------------------------------------------------------------
__global__ void wswz_kernel(const float* __restrict__ W, _Float16* __restrict__ out,
                            int Kin, int Kpad, int N) {
  int i = blockIdx.x * blockDim.x + threadIdx.x;
  if (i >= Kpad * N) return;
  int e = i & 15;            // element within lane
  int lane = (i >> 4) & 31;
  int t = i >> 9;            // (kb*NT + tn)
  int NT = N >> 4;
  int kb = t / NT, tn = t - kb * NT;
  int k = kb * 32 + ((lane >> 4) << 3) + (e & 7) + ((e >> 3) << 4);
  int n = tn * 16 + (lane & 15);
  out[i] = (k < Kin) ? (_Float16)W[(size_t)k * N + n] : (_Float16)0.f;
}

// ---------------------------------------------------------------------------
// Farthest point sampling: one workgroup per batch, dmin resident in LDS
// ---------------------------------------------------------------------------
__global__ void __launch_bounds__(256) fps_kernel(const float* __restrict__ xyz,
                                                  int N, int S, int* __restrict__ idx) {
  __shared__ float dmin[8192];
  __shared__ float rv[256];
  __shared__ int   ri[256];
  __shared__ float cc[3];
  __shared__ int   sfar;
  const int b = blockIdx.x, tid = threadIdx.x;
  const float* p = xyz + (size_t)b * N * 3;
  int* o = idx + (size_t)b * S;
  for (int j = tid; j < N; j += 256) dmin[j] = 1e10f;
  if (tid == 0) sfar = 0;
  __syncthreads();
  for (int st = 0; st < S; ++st) {
    if (tid == 0) {
      int f = sfar;
      o[st] = f;
      cc[0] = p[f * 3 + 0]; cc[1] = p[f * 3 + 1]; cc[2] = p[f * 3 + 2];
    }
    __syncthreads();
    float bx = cc[0], by = cc[1], bz = cc[2];
    float bestv = -1.f; int besti = 0;
    for (int j = tid; j < N; j += 256) {
      float dx = p[j * 3 + 0] - bx, dy = p[j * 3 + 1] - by, dz = p[j * 3 + 2] - bz;
      float d = dx * dx + dy * dy + dz * dz;
      float dm = fminf(dmin[j], d);
      dmin[j] = dm;
      if (dm > bestv) { bestv = dm; besti = j; }
    }
    rv[tid] = bestv; ri[tid] = besti;
    __syncthreads();
    for (int off = 128; off; off >>= 1) {
      if (tid < off) {
        float ov = rv[tid + off]; int oi = ri[tid + off];
        if (ov > rv[tid] || (ov == rv[tid] && oi < ri[tid])) { rv[tid] = ov; ri[tid] = oi; }
      }
      __syncthreads();
    }
    if (tid == 0) sfar = ri[0];
    __syncthreads();
  }
}

__global__ void gather_kernel(const float* __restrict__ xyz, const int* __restrict__ idx,
                              float* __restrict__ out, int N, int S) {
  int i = blockIdx.x * blockDim.x + threadIdx.x;
  if (i >= 16 * S) return;
  int b = i / S;
  int n = idx[i];
  const float* pp = xyz + ((size_t)b * N + n) * 3;
  out[(size_t)i * 3 + 0] = pp[0];
  out[(size_t)i * 3 + 1] = pp[1];
  out[(size_t)i * 3 + 2] = pp[2];
}

// ---------------------------------------------------------------------------
// KNN: one wave32 per query; per-lane sorted top-K (dist|idx packed u64),
// then K rounds of wave-min extraction via lane shuffles.
// ---------------------------------------------------------------------------
__device__ inline unsigned long long shflxor_u64(unsigned long long v, int mask) {
  int lo = (int)(v & 0xffffffffULL);
  int hi = (int)(v >> 32);
  lo = __shfl_xor(lo, mask, 32);
  hi = __shfl_xor(hi, mask, 32);
  return ((unsigned long long)(unsigned int)hi << 32) | (unsigned int)lo;
}

template <int K>
__global__ void __launch_bounds__(32) knn_kernel(const float* __restrict__ xyz,
                                                 const float* __restrict__ q,
                                                 int N, int S, int* __restrict__ out) {
  const int bs = blockIdx.x;
  const int b = bs / S;
  const int lane = threadIdx.x;
  const float* pts = xyz + (size_t)b * N * 3;
  const float qx = q[(size_t)bs * 3 + 0], qy = q[(size_t)bs * 3 + 1], qz = q[(size_t)bs * 3 + 2];
  unsigned long long lst[K];
#pragma unroll
  for (int i = 0; i < K; ++i) lst[i] = ~0ULL;
  for (int j = lane; j < N; j += 32) {
    float dx = pts[j * 3 + 0] - qx, dy = pts[j * 3 + 1] - qy, dz = pts[j * 3 + 2] - qz;
    float d = dx * dx + dy * dy + dz * dz;
    unsigned long long key = ((unsigned long long)__float_as_uint(d) << 32) | (unsigned int)j;
    if (key < lst[K - 1]) {
      lst[K - 1] = key;
#pragma unroll
      for (int i = K - 1; i > 0; --i) {
        if (lst[i] < lst[i - 1]) { unsigned long long t = lst[i]; lst[i] = lst[i - 1]; lst[i - 1] = t; }
      }
    }
  }
  for (int k = 0; k < K; ++k) {
    unsigned long long head = lst[0];
    unsigned long long mn = head;
#pragma unroll
    for (int off = 16; off; off >>= 1) {
      unsigned long long o = shflxor_u64(mn, off);
      if (o < mn) mn = o;
    }
    if (lane == 0) out[(size_t)bs * K + k] = (int)(mn & 0xffffffffULL);
    if (head == mn) {
#pragma unroll
      for (int i = 0; i < K - 1; ++i) lst[i] = lst[i + 1];
      lst[K - 1] = ~0ULL;
    }
  }
}

// ---------------------------------------------------------------------------
// Grouping: build padded f16 activation rows for the MLP GEMMs
// ---------------------------------------------------------------------------
__global__ void group1_kernel(const float* __restrict__ x, const float* __restrict__ nx,
                              const int* __restrict__ knn, _Float16* __restrict__ out) {
  int i = blockIdx.x * blockDim.x + threadIdx.x;     // row = ((b*1024+s)*16+k)
  if (i >= 16 * 1024 * 16) return;
  int bs = i >> 4;
  int b = bs >> 10;
  int n = knn[i];
  const float* p = x + ((size_t)b * 8192 + n) * 3;
  const float* qq = nx + (size_t)bs * 3;
  _Float16* o = out + (size_t)i * 32;
  o[0] = (_Float16)(p[0] - qq[0]);
  o[1] = (_Float16)(p[1] - qq[1]);
  o[2] = (_Float16)(p[2] - qq[2]);
#pragma unroll
  for (int c = 3; c < 32; ++c) o[c] = (_Float16)0.f;
}

__global__ void group2_kernel(const float* __restrict__ xyz1, const float* __restrict__ nx2,
                              const int* __restrict__ knn, const float* __restrict__ p1,
                              _Float16* __restrict__ out) {
  int i = blockIdx.x * blockDim.x + threadIdx.x;     // row = ((b*256+s)*32+k)
  if (i >= 16 * 256 * 32) return;
  int bs = i >> 5;
  int b = bs >> 8;
  int n = knn[i];
  const float* p = xyz1 + ((size_t)b * 1024 + n) * 3;
  const float* qq = nx2 + (size_t)bs * 3;
  _Float16* o = out + (size_t)i * 96;
  o[0] = (_Float16)(p[0] - qq[0]);
  o[1] = (_Float16)(p[1] - qq[1]);
  o[2] = (_Float16)(p[2] - qq[2]);
  const float* pf = p1 + (size_t)b * 64 * 1024 + n;  // p1[b][c][n], stride 1024
#pragma unroll 4
  for (int c = 0; c < 64; ++c) o[3 + c] = (_Float16)pf[(size_t)c * 1024];
#pragma unroll
  for (int c = 67; c < 96; ++c) o[c] = (_Float16)0.f;
}

// ---------------------------------------------------------------------------
// WMMA GEMM: one wave computes a 16 x N row-stripe (NT = N/16 column tiles).
// A fragment is loaded once per k-step and reused for all NT wmma ops.
// B comes from the pre-swizzled buffer: one 32B contiguous load per (k-step, tn).
//   A lane layout (16-bit A 16x32): lane<16 K{0..7,16..23}, lane>=16 K{8..15,24..31}
//   D layout: VGPR r, lane l -> M = r + 8*(l>=16), Ncol = l&15
// ---------------------------------------------------------------------------
template <int NT>
__global__ void __launch_bounds__(128) wmma_gemm_kernel(const _Float16* __restrict__ A,
                                                        const _Float16* __restrict__ Bswz,
                                                        _Float16* __restrict__ Out,
                                                        int rows, int Kpad) {
  const int lane = threadIdx.x & 31;
  const int wv = threadIdx.x >> 5;
  const int rowtiles = rows >> 4;
  const int tm = blockIdx.x * 4 + wv;            // wave-uniform
  if (tm >= rowtiles) return;
  const int N = NT * 16;
  const int upper = lane >> 4;                   // 0 or 1
  const int l15 = lane & 15;
  v8f acc[NT];
#pragma unroll
  for (int t = 0; t < NT; ++t) acc[t] = (v8f){};
  const _Float16* arow = A + (size_t)(tm * 16 + l15) * Kpad + (upper ? 8 : 0);
  const int ksteps = Kpad >> 5;
  for (int kb = 0; kb < ksteps; ++kb) {
    v8h alo = *(const v8h*)(arow + kb * 32);
    v8h ahi = *(const v8h*)(arow + kb * 32 + 16);
    v16h av = __builtin_shufflevector(alo, ahi, 0, 1, 2, 3, 4, 5, 6, 7,
                                      8, 9, 10, 11, 12, 13, 14, 15);
    const _Float16* bbase = Bswz + (((size_t)kb * NT) * 32 + lane) * 16;
#pragma unroll
    for (int t = 0; t < NT; ++t) {
      v16h bv = *(const v16h*)(bbase + (size_t)t * 32 * 16);
      acc[t] = __builtin_amdgcn_wmma_f32_16x16x32_f16(false, av, false, bv,
                                                      (short)0, acc[t], false, false);
    }
  }
  _Float16* orow = Out + (size_t)(tm * 16 + (upper << 3)) * N + l15;
#pragma unroll
  for (int t = 0; t < NT; ++t) {
#pragma unroll
    for (int r = 0; r < 8; ++r) orow[(size_t)r * N + t * 16] = (_Float16)acc[t][r];
  }
}

// ---------------------------------------------------------------------------
// Training-mode BN: per-channel mean/istd over all rows, then scale+shift+ReLU
// ---------------------------------------------------------------------------
__global__ void __launch_bounds__(256) bn_stats_kernel(const _Float16* __restrict__ act,
                                                       int rows, int C,
                                                       float* __restrict__ mean,
                                                       float* __restrict__ istd) {
  __shared__ float ss[256], sq[256];
  const int c = blockIdx.x, tid = threadIdx.x;
  float s = 0.f, s2 = 0.f;
  for (int r = tid; r < rows; r += 256) {
    float v = (float)act[(size_t)r * C + c];
    s += v; s2 += v * v;
  }
  ss[tid] = s; sq[tid] = s2;
  __syncthreads();
  for (int o = 128; o; o >>= 1) {
    if (tid < o) { ss[tid] += ss[tid + o]; sq[tid] += sq[tid + o]; }
    __syncthreads();
  }
  if (tid == 0) {
    float mu = ss[0] / (float)rows;
    float var = sq[0] / (float)rows - mu * mu;
    mean[c] = mu;
    istd[c] = rsqrtf(var + 1e-5f);
  }
}

__global__ void bn_relu_kernel(_Float16* __restrict__ act, int n, int C,
                               const float* __restrict__ gamma, const float* __restrict__ beta,
                               const float* __restrict__ mean, const float* __restrict__ istd) {
  int i = blockIdx.x * blockDim.x + threadIdx.x;
  if (i >= n) return;
  int c = i % C;
  float v = (float)act[i];
  v = gamma[c] * (v - mean[c]) * istd[c] + beta[c];
  act[i] = (_Float16)fmaxf(v, 0.f);
}

// ---------------------------------------------------------------------------
// Max over K neighbors -> (B, C, S) fp32; then max over S -> g (B, C)
// ---------------------------------------------------------------------------
__global__ void maxk_kernel(const _Float16* __restrict__ act, float* __restrict__ out,
                            int S, int K) {
  const int bs = blockIdx.x, C = blockDim.x, c = threadIdx.x;
  const int b = bs / S, s = bs - b * S;
  const _Float16* base = act + (size_t)bs * K * C + c;
  float m = -3.0e38f;
  for (int k = 0; k < K; ++k) m = fmaxf(m, (float)base[(size_t)k * C]);
  out[((size_t)b * C + c) * S + s] = m;
}

__global__ void gmax_kernel(const float* __restrict__ p2, float* __restrict__ g) {
  const int b = blockIdx.x, c = threadIdx.x;
  const float* base = p2 + ((size_t)b * 256 + c) * 256;
  float m = -3.0e38f;
  for (int s = 0; s < 256; ++s) m = fmaxf(m, base[s]);
  g[b * 256 + c] = m;
}

// ---------------------------------------------------------------------------
// Head: gconv (BN over batch axis) + SE gate; single workgroup.
// ---------------------------------------------------------------------------
__global__ void __launch_bounds__(256) head_kernel(
    const float* __restrict__ g, const float* __restrict__ W1, const float* __restrict__ g1,
    const float* __restrict__ b1, const float* __restrict__ W2, const float* __restrict__ b2,
    const float* __restrict__ f1w, const float* __restrict__ f1b,
    const float* __restrict__ f2w, const float* __restrict__ f2b,
    float* __restrict__ yout, float* __restrict__ gsmall) {
  __shared__ float gsh[16 * 256];
  __shared__ float hsh[16 * 256];
  __shared__ float ysh[16 * 256];
  __shared__ float z1sh[16 * 32];
  const int d = threadIdx.x;
  for (int i = d; i < 4096; i += 256) gsh[i] = g[i];
  __syncthreads();
  float t[16];
#pragma unroll
  for (int b = 0; b < 16; ++b) t[b] = 0.f;
  for (int c = 0; c < 256; ++c) {
    float w = W1[c * 256 + d];
#pragma unroll
    for (int b = 0; b < 16; ++b) t[b] += gsh[b * 256 + c] * w;
  }
  float mu = 0.f;
#pragma unroll
  for (int b = 0; b < 16; ++b) mu += t[b];
  mu *= (1.f / 16.f);
  float var = 0.f;
#pragma unroll
  for (int b = 0; b < 16; ++b) { float dv = t[b] - mu; var += dv * dv; }
  var *= (1.f / 16.f);
  float is = rsqrtf(var + 1e-5f);
  float ga = g1[d], be = b1[d];
#pragma unroll
  for (int b = 0; b < 16; ++b) hsh[b * 256 + d] = fmaxf(ga * (t[b] - mu) * is + be, 0.f);
  __syncthreads();
  float y[16];
  float bb = b2[d];
#pragma unroll
  for (int b = 0; b < 16; ++b) y[b] = bb;
  for (int c = 0; c < 256; ++c) {
    float w = W2[c * 256 + d];
#pragma unroll
    for (int b = 0; b < 16; ++b) y[b] += hsh[b * 256 + c] * w;
  }
#pragma unroll
  for (int b = 0; b < 16; ++b) ysh[b * 256 + d] = y[b];
  __syncthreads();
  if (d < 32) {
#pragma unroll
    for (int b = 0; b < 16; ++b) {
      float z = f1b[d];
      for (int c = 0; c < 256; ++c) z += ysh[b * 256 + c] * f1w[c * 32 + d];
      z1sh[b * 32 + d] = fmaxf(z, 0.f);
    }
  }
  __syncthreads();
#pragma unroll
  for (int b = 0; b < 16; ++b) {
    float s = f2b[d];
#pragma unroll
    for (int j = 0; j < 32; ++j) s += z1sh[b * 32 + j] * f2w[j * 256 + d];
    float a = 1.f / (1.f + expf(-s));
    yout[b * 256 + d] = y[b] * a;
  }
  if (d < 128) {
#pragma unroll
    for (int b = 0; b < 16; ++b) gsmall[b * 128 + d] = gsh[b * 256 + d];
  }
}

// ---------------------------------------------------------------------------
// Host orchestration
// ---------------------------------------------------------------------------
extern "C" void kernel_launch(void* const* d_in, const int* in_sizes, int n_in,
                              void* d_out, int out_size, void* d_ws, size_t ws_size,
                              hipStream_t stream) {
  (void)in_sizes; (void)n_in; (void)out_size; (void)ws_size;
  const float* x = (const float*)d_in[0];
  const float* sa1W[3] = {(const float*)d_in[1], (const float*)d_in[4], (const float*)d_in[7]};
  const float* sa1G[3] = {(const float*)d_in[2], (const float*)d_in[5], (const float*)d_in[8]};
  const float* sa1B[3] = {(const float*)d_in[3], (const float*)d_in[6], (const float*)d_in[9]};
  const float* sa2W[3] = {(const float*)d_in[10], (const float*)d_in[13], (const float*)d_in[16]};
  const float* sa2G[3] = {(const float*)d_in[11], (const float*)d_in[14], (const float*)d_in[17]};
  const float* sa2B[3] = {(const float*)d_in[12], (const float*)d_in[15], (const float*)d_in[18]};

  float* out = (float*)d_out;
  float* y_out = out;                           // (16,256)
  float* p1_out = out + 4096;                   // (16,64,1024)
  float* p2_out = out + 4096 + 1048576;         // (16,256,256)
  float* gsmall_out = out + 4096 + 2 * 1048576; // (16,128)

  uintptr_t cur = (uintptr_t)d_ws;
  auto take = [&](size_t bytes) {
    uintptr_t p = (cur + 255) & ~(uintptr_t)255;
    cur = p + bytes;
    return (void*)p;
  };
  _Float16* w16[6];
  const int wkin[6]  = {3, 32, 32, 67, 128, 128};
  const int wkpad[6] = {32, 32, 32, 96, 128, 128};
  const int wn[6]    = {32, 32, 64, 128, 128, 256};
  for (int i = 0; i < 6; ++i) w16[i] = (_Float16*)take((size_t)wkpad[i] * wn[i] * 2);
  int* fpsidx1 = (int*)take((size_t)16 * 1024 * 4);
  int* fpsidx2 = (int*)take((size_t)16 * 256 * 4);
  float* nx1 = (float*)take((size_t)16 * 1024 * 3 * 4);
  float* nx2 = (float*)take((size_t)16 * 256 * 3 * 4);
  int* knn1 = (int*)take((size_t)16 * 1024 * 16 * 4);
  int* knn2 = (int*)take((size_t)16 * 256 * 32 * 4);
  float* meanb = (float*)take(256 * 4);
  float* istdb = (float*)take(256 * 4);
  float* gbuf = (float*)take((size_t)16 * 256 * 4);
  _Float16* bufA = (_Float16*)take((size_t)16777216 * 2);  // 33.5 MB
  _Float16* bufB = (_Float16*)take((size_t)33554432 * 2);  // 67 MB

  // weight conversion into pre-swizzled WMMA B layout
  const float* wsrc[6] = {sa1W[0], sa1W[1], sa1W[2], sa2W[0], sa2W[1], sa2W[2]};
  for (int i = 0; i < 6; ++i) {
    int n = wkpad[i] * wn[i];
    wswz_kernel<<<(n + 255) / 256, 256, 0, stream>>>(wsrc[i], w16[i], wkin[i], wkpad[i], wn[i]);
  }

  auto gemm = [&](const _Float16* A, const _Float16* Bw, _Float16* O,
                  int rows, int Kpad, int N) {
    int blocks = (rows / 16 + 3) / 4;
    switch (N >> 4) {
      case 2:  wmma_gemm_kernel<2> <<<blocks, 128, 0, stream>>>(A, Bw, O, rows, Kpad); break;
      case 4:  wmma_gemm_kernel<4> <<<blocks, 128, 0, stream>>>(A, Bw, O, rows, Kpad); break;
      case 8:  wmma_gemm_kernel<8> <<<blocks, 128, 0, stream>>>(A, Bw, O, rows, Kpad); break;
      default: wmma_gemm_kernel<16><<<blocks, 128, 0, stream>>>(A, Bw, O, rows, Kpad); break;
    }
  };
  auto bn = [&](_Float16* act, int rows, int C, const float* ga, const float* be) {
    bn_stats_kernel<<<C, 256, 0, stream>>>(act, rows, C, meanb, istdb);
    int n = rows * C;
    bn_relu_kernel<<<(n + 255) / 256, 256, 0, stream>>>(act, n, C, ga, be, meanb, istdb);
  };

  // ---------------- SA1: N=8192 -> S=1024, K=16, 3->32->32->64 ----------------
  fps_kernel<<<16, 256, 0, stream>>>(x, 8192, 1024, fpsidx1);
  gather_kernel<<<(16 * 1024 + 255) / 256, 256, 0, stream>>>(x, fpsidx1, nx1, 8192, 1024);
  knn_kernel<16><<<16 * 1024, 32, 0, stream>>>(x, nx1, 8192, 1024, knn1);
  group1_kernel<<<(262144 + 255) / 256, 256, 0, stream>>>(x, nx1, knn1, bufA);
  gemm(bufA, w16[0], bufB, 262144, 32, 32);  bn(bufB, 262144, 32, sa1G[0], sa1B[0]);
  gemm(bufB, w16[1], bufA, 262144, 32, 32);  bn(bufA, 262144, 32, sa1G[1], sa1B[1]);
  gemm(bufA, w16[2], bufB, 262144, 32, 64);  bn(bufB, 262144, 64, sa1G[2], sa1B[2]);
  maxk_kernel<<<16 * 1024, 64, 0, stream>>>(bufB, p1_out, 1024, 16);

  // ---------------- SA2: N=1024 -> S=256, K=32, 67->128->128->256 -------------
  fps_kernel<<<16, 256, 0, stream>>>(nx1, 1024, 256, fpsidx2);
  gather_kernel<<<(16 * 256 + 255) / 256, 256, 0, stream>>>(nx1, fpsidx2, nx2, 1024, 256);
  knn_kernel<32><<<16 * 256, 32, 0, stream>>>(nx1, nx2, 1024, 256, knn2);
  group2_kernel<<<(131072 + 255) / 256, 256, 0, stream>>>(nx1, nx2, knn2, p1_out, bufA);
  gemm(bufA, w16[3], bufB, 131072, 96, 128);  bn(bufB, 131072, 128, sa2G[0], sa2B[0]);
  gemm(bufB, w16[4], bufA, 131072, 128, 128); bn(bufA, 131072, 128, sa2G[1], sa2B[1]);
  gemm(bufA, w16[5], bufB, 131072, 128, 256); bn(bufB, 131072, 256, sa2G[2], sa2B[2]);
  maxk_kernel<<<16 * 256, 256, 0, stream>>>(bufB, p2_out, 256, 32);

  // ---------------- global pooling + head ------------------------------------
  gmax_kernel<<<16, 256, 0, stream>>>(p2_out, gbuf);
  head_kernel<<<1, 256, 0, stream>>>(gbuf,
      (const float*)d_in[19], (const float*)d_in[20], (const float*)d_in[21],
      (const float*)d_in[22], (const float*)d_in[23],
      (const float*)d_in[24], (const float*)d_in[25],
      (const float*)d_in[26], (const float*)d_in[27],
      y_out, gsmall_out);
}
